// TransformerBlock_26121991094960
// MI455X (gfx1250) — matrix operand order repaired and checked by reading, not compiled, not run on hardware
//
#include <hip/hip_runtime.h>
#include <cmath>

// ---------------------------------------------------------------------------
// MI455X (gfx1250) transformer block, bf16 WMMA pipeline.
//   features[8,2048,64] -> fc1 -> x -> Q,K,V -> softmax(QK^T/sqrt(d)) V -> fc2+relu
// All GEMMs run on v_wmma_f32_16x16x32_bf16 (f32 accumulate).
// Attention is a single-pass flash kernel (online softmax, cooperative waves).
// ---------------------------------------------------------------------------

typedef __attribute__((ext_vector_type(16))) __bf16 bf16x16;
typedef __attribute__((ext_vector_type(8)))  __bf16 bf16x8;
typedef __attribute__((ext_vector_type(8)))  float  f32x8;

__device__ __forceinline__ f32x8 wmma_bf16(bf16x16 a, bf16x16 b, f32x8 c) {
  // (neg_a, A, neg_b, B, c_mod, C, reuse_a, reuse_b)
  return __builtin_amdgcn_wmma_f32_16x16x32_bf16(false, a, false, b, (short)0, c,
                                                 false, false);
}

// A operand: 16x32 bf16 tile from row-major [16 x ld] at column offset k0.
// Lane L holds row (L&15); halves: e<8 -> k0+half*8+e, e>=8 -> k0+16+half*8+(e-8).
__device__ __forceinline__ bf16x16 load_a_tile(const __bf16* base, int ld,
                                               int lane, int k0) {
  int m = lane & 15, half = lane >> 4;
  const __bf16* p = base + (size_t)m * ld + k0 + half * 8;
  union { bf16x16 v; bf16x8 h[2]; } u;
  u.h[0] = *(const bf16x8*)(p);
  u.h[1] = *(const bf16x8*)(p + 16);
  return u.v;
}

// B operand: 32x16 bf16 tile. Memory stores B transposed row-major: src[n][k],
// leading dim ld. Lane L holds column n0+(L&15); halves: e<8 -> k0+half*8+e, e>=8 -> k0+16+half*8+(e-8).
__device__ __forceinline__ bf16x16 load_b_tile(const __bf16* base, int ld,
                                               int lane, int n0, int k0) {
  int n = lane & 15, half = lane >> 4;
  const __bf16* p = base + (size_t)(n0 + n) * ld + k0 + half * 8;
  union { bf16x16 v; bf16x8 h[2]; } u;
  u.h[0] = *(const bf16x8*)(p);
  u.h[1] = *(const bf16x8*)(p + 16);
  return u.v;
}

// ------------------------------ prep kernels -------------------------------

__global__ void k_f32_to_bf16(const float* __restrict__ src,
                              __bf16* __restrict__ dst, int n) {
  int i = blockIdx.x * blockDim.x + threadIdx.x;
  if (i < n) dst[i] = (__bf16)src[i];
}

// src[K][N] (fp32, row-major) -> dst[N][K] (bf16): WMMA-B-ready weight layout.
__global__ void k_transpose_to_bf16(const float* __restrict__ src,
                                    __bf16* __restrict__ dst, int K, int N) {
  int i = blockIdx.x * blockDim.x + threadIdx.x;
  if (i < K * N) {
    int n = i / K, kk = i % K;
    dst[i] = (__bf16)src[(size_t)kk * N + n];
  }
}

// --------------------------- generic bf16 GEMM -----------------------------
// One wave per 32x64 C tile (2 A-tiles share 4 B-tiles -> 1.5 loads/WMMA).
// A: [M][K] bf16 row-major. Bt: [N][K] bf16.
// Epilogue: *scale, +bias, optional relu; writes bf16, f32, or bf16-transposed
// (per-batch [N][tokens], used for V).
__global__ void __launch_bounds__(32)
k_gemm_bf16(const __bf16* __restrict__ A, const __bf16* __restrict__ Bt,
            const float* __restrict__ bias, float scale, int M, int K, int N,
            __bf16* __restrict__ outB, float* __restrict__ outF,
            __bf16* __restrict__ outT, int tokens, int relu) {
  int lane = threadIdx.x & 31;
  int m0 = blockIdx.x * 32;
  int n0 = blockIdx.y * 64;
  int half = lane >> 4, nl = lane & 15;
  const __bf16* A0 = A + (size_t)m0 * K;
  const __bf16* A1 = A0 + (size_t)16 * K;

  f32x8 acc[2][4] = {};
  for (int k0 = 0; k0 < K; k0 += 32) {
    bf16x16 a0 = load_a_tile(A0, K, lane, k0);
    bf16x16 a1 = load_a_tile(A1, K, lane, k0);
#pragma unroll
    for (int s = 0; s < 4; ++s) {
      bf16x16 b = load_b_tile(Bt, K, lane, n0 + s * 16, k0);
      acc[0][s] = wmma_bf16(a0, b, acc[0][s]);
      acc[1][s] = wmma_bf16(a1, b, acc[1][s]);
    }
  }

#pragma unroll
  for (int mt = 0; mt < 2; ++mt) {
    int mbase = m0 + mt * 16;
#pragma unroll
    for (int s = 0; s < 4; ++s) {
      int col = n0 + s * 16 + nl;
      float bi = bias ? bias[col] : 0.0f;
      if (outT) {
        int bb = mbase / tokens;
        int tok0 = (mbase % tokens) + half * 8;
        union { bf16x8 v; __bf16 e[8]; } pk;
#pragma unroll
        for (int r = 0; r < 8; ++r) pk.e[r] = (__bf16)(acc[mt][s][r] * scale + bi);
        *(bf16x8*)(outT + (size_t)bb * N * tokens + (size_t)col * tokens + tok0) =
            pk.v;
      } else {
#pragma unroll
        for (int r = 0; r < 8; ++r) {
          int row = mbase + half * 8 + r;
          float v = acc[mt][s][r] * scale + bi;
          if (relu) v = fmaxf(v, 0.0f);
          if (outB) outB[(size_t)row * N + col] = (__bf16)v;
          if (outF) outF[(size_t)row * N + col] = v;
        }
      }
    }
  }
}

// --------------------- fused flash attention (1 pass) ----------------------
// Block = 8 waves = 2 query tiles x 4 cooperating waves. Per 128-key block:
//  phase 1: each wave computes S (QK^T) for its 32-key slice (WMMA), reduces
//           per-row slice max, publishes it to LDS.
//  phase 2: all 4 waves agree on m_new = max(m_run, block max); write
//           P = exp2(S - m_new) (bf16) into a shared LDS slab; publish slice
//           row sums.
//  phase 3: l_run = l_run*exp2(m_old-m_new) + block_sum; rescale private O
//           chunk; accumulate O += P @ V from the LDS slab (WMMA).
// Final: res = O / l_run  (q was pre-scaled by log2(e)/sqrt(d) -> exp2 math).
__global__ void __launch_bounds__(256)
k_attn_flash(const __bf16* __restrict__ q, const __bf16* __restrict__ kmat,
             const __bf16* __restrict__ vT, __bf16* __restrict__ res,
             int tokens, int dmodel) {
  __shared__ __bf16 Ps[2][16][144];  // 128 keys + pad to 32B rows
  __shared__ float MxS[2][4][16];    // per-slice row max
  __shared__ float SmS[2][4][16];    // per-slice row sum
  int lane = threadIdx.x & 31;
  int wave = threadIdx.x >> 5;
  int qt = wave >> 2;  // query tile within block
  int dc = wave & 3;   // 32-key slice owner / 128-wide d-chunk owner
  int tile = blockIdx.x * 2 + qt;
  int qm0 = tile * 16;
  int b = qm0 / tokens;
  int half = lane >> 4, nl = lane & 15;
  const __bf16* qb = q + (size_t)qm0 * dmodel;
  const __bf16* kb = kmat + (size_t)b * tokens * dmodel;
  const __bf16* vb = vT + (size_t)b * dmodel * tokens;

  float mrun[8], lrun[8];
#pragma unroll
  for (int r = 0; r < 8; ++r) { mrun[r] = -3.0e38f; lrun[r] = 0.0f; }
  f32x8 o[8] = {};

  for (int kblk = 0; kblk < tokens; kblk += 128) {
    int kq = kblk + dc * 32;
    if (kblk + 128 < tokens)
      __builtin_prefetch(kb + (size_t)(kblk + 128 + dc * 32) * dmodel, 0, 0);

    // ---- phase 1: S = (scaled Q) K^T for this wave's 32-key slice ----
    f32x8 s0 = {}, s1 = {};
    for (int k0 = 0; k0 < dmodel; k0 += 32) {
      bf16x16 a  = load_a_tile(qb, dmodel, lane, k0);
      bf16x16 b0 = load_b_tile(kb, dmodel, lane, kq, k0);
      bf16x16 b1 = load_b_tile(kb, dmodel, lane, kq + 16, k0);
      s0 = wmma_bf16(a, b0, s0);
      s1 = wmma_bf16(a, b1, s1);
    }
    float mx[8];
#pragma unroll
    for (int r = 0; r < 8; ++r) {
      float v = fmaxf(s0[r], s1[r]);
#pragma unroll
      for (int off = 8; off >= 1; off >>= 1)
        v = fmaxf(v, __shfl_xor(v, off, 32));  // reduce within 16-lane half
      mx[r] = v;
    }
    if (nl == 0) {
#pragma unroll
      for (int r = 0; r < 8; ++r) MxS[qt][dc][half * 8 + r] = mx[r];
    }
    __syncthreads();

    // ---- phase 2: agree on m_new, emit P, publish slice sums ----
    float mnew[8], ssum[8];
#pragma unroll
    for (int r = 0; r < 8; ++r) {
      int row = half * 8 + r;
      float bm = fmaxf(fmaxf(MxS[qt][0][row], MxS[qt][1][row]),
                       fmaxf(MxS[qt][2][row], MxS[qt][3][row]));
      mnew[r] = fmaxf(mrun[r], bm);
      float p0 = exp2f(s0[r] - mnew[r]);
      float p1 = exp2f(s1[r] - mnew[r]);
      Ps[qt][row][dc * 32 + nl]      = (__bf16)p0;
      Ps[qt][row][dc * 32 + 16 + nl] = (__bf16)p1;
      float sm = p0 + p1;
#pragma unroll
      for (int off = 8; off >= 1; off >>= 1) sm += __shfl_xor(sm, off, 32);
      ssum[r] = sm;
    }
    if (nl == 0) {
#pragma unroll
      for (int r = 0; r < 8; ++r) SmS[qt][dc][half * 8 + r] = ssum[r];
    }
    __syncthreads();

    // ---- phase 3: update (m,l), rescale O, accumulate P @ V ----
#pragma unroll
    for (int r = 0; r < 8; ++r) {
      int row = half * 8 + r;
      float bs = (SmS[qt][0][row] + SmS[qt][1][row]) +
                 (SmS[qt][2][row] + SmS[qt][3][row]);
      float corr = exp2f(mrun[r] - mnew[r]);
      lrun[r] = lrun[r] * corr + bs;
      mrun[r] = mnew[r];
#pragma unroll
      for (int ns = 0; ns < 8; ++ns) o[ns][r] = o[ns][r] * corr;
    }
#pragma unroll
    for (int k0 = 0; k0 < 128; k0 += 32) {
      bf16x16 a = load_a_tile(&Ps[qt][0][0], 144, lane, k0);
#pragma unroll
      for (int ns = 0; ns < 8; ++ns) {
        bf16x16 bb = load_b_tile(vb, tokens, lane, dc * 128 + ns * 16, kblk + k0);
        o[ns] = wmma_bf16(a, bb, o[ns]);
      }
    }
    __syncthreads();  // protect Ps/MxS before next block overwrites
  }

  float linv[8];
#pragma unroll
  for (int r = 0; r < 8; ++r) linv[r] = 1.0f / lrun[r];
#pragma unroll
  for (int ns = 0; ns < 8; ++ns) {
#pragma unroll
    for (int r = 0; r < 8; ++r) {
      int row = qm0 + half * 8 + r;
      int col = dc * 128 + ns * 16 + nl;
      res[(size_t)row * dmodel + col] = (__bf16)(o[ns][r] * linv[r]);
    }
  }
}

// ------------------------------- launcher ----------------------------------

extern "C" void kernel_launch(void* const* d_in, const int* in_sizes, int n_in,
                              void* d_out, int out_size, void* d_ws,
                              size_t ws_size, hipStream_t stream) {
  const float* features = (const float*)d_in[0];
  const float* W1 = (const float*)d_in[1];
  const float* b1 = (const float*)d_in[2];
  const float* Wq = (const float*)d_in[3];
  const float* Wk = (const float*)d_in[4];
  const float* Wv = (const float*)d_in[5];
  const float* W2 = (const float*)d_in[6];
  const float* b2 = (const float*)d_in[7];
  float* out = (float*)d_out;

  const int B = 8, T = 2048, D = 512, P = 64;
  const int M = B * T;  // 16384

  // workspace carve-out (~68 MB)
  char* w = (char*)d_ws;
  auto carve = [&](size_t bytes) {
    char* p = w;
    w += (bytes + 255) & ~(size_t)255;
    return p;
  };
  __bf16* featbf = (__bf16*)carve((size_t)M * P * 2);
  __bf16* W1t    = (__bf16*)carve((size_t)P * D * 2);
  __bf16* Wqt    = (__bf16*)carve((size_t)D * D * 2);
  __bf16* Wkt    = (__bf16*)carve((size_t)D * D * 2);
  __bf16* Wvt    = (__bf16*)carve((size_t)D * D * 2);
  __bf16* W2t    = (__bf16*)carve((size_t)D * P * 2);
  __bf16* x_bf   = (__bf16*)carve((size_t)M * D * 2);
  __bf16* q_bf   = (__bf16*)carve((size_t)M * D * 2);
  __bf16* k_bf   = (__bf16*)carve((size_t)M * D * 2);
  __bf16* vT_bf  = (__bf16*)carve((size_t)M * D * 2);  // per-batch [D][T]
  __bf16* res_bf = x_bf;  // x no longer needed after QKV -> reuse for attn out

  // prep: convert + transpose weights to WMMA-B layout
  int nfeat = M * P;
  k_f32_to_bf16<<<(nfeat + 255) / 256, 256, 0, stream>>>(features, featbf, nfeat);
  k_transpose_to_bf16<<<(P * D + 255) / 256, 256, 0, stream>>>(W1, W1t, P, D);
  k_transpose_to_bf16<<<(D * D + 255) / 256, 256, 0, stream>>>(Wq, Wqt, D, D);
  k_transpose_to_bf16<<<(D * D + 255) / 256, 256, 0, stream>>>(Wk, Wkt, D, D);
  k_transpose_to_bf16<<<(D * D + 255) / 256, 256, 0, stream>>>(Wv, Wvt, D, D);
  k_transpose_to_bf16<<<(D * P + 255) / 256, 256, 0, stream>>>(W2, W2t, D, P);

  // fc1: x = features @ W1 + b1   [M,64]x[64,512]
  k_gemm_bf16<<<dim3(M / 32, D / 64), 32, 0, stream>>>(
      featbf, W1t, b1, 1.0f, M, P, D, x_bf, nullptr, nullptr, T, 0);

  // q (folds 1/sqrt(D) and log2(e) so softmax can use exp2), k, v (transposed)
  float qscale = 1.44269504088896341f / sqrtf((float)D);
  k_gemm_bf16<<<dim3(M / 32, D / 64), 32, 0, stream>>>(
      x_bf, Wqt, nullptr, qscale, M, D, D, q_bf, nullptr, nullptr, T, 0);
  k_gemm_bf16<<<dim3(M / 32, D / 64), 32, 0, stream>>>(
      x_bf, Wkt, nullptr, 1.0f, M, D, D, k_bf, nullptr, nullptr, T, 0);
  k_gemm_bf16<<<dim3(M / 32, D / 64), 32, 0, stream>>>(
      x_bf, Wvt, nullptr, 1.0f, M, D, D, nullptr, nullptr, vT_bf, T, 0);

  // fused flash attention: res = softmax(QK^T)V
  k_attn_flash<<<dim3(M / 16 / 2), 256, 0, stream>>>(q_bf, k_bf, vT_bf, res_bf,
                                                     T, D);

  // fc2 + relu -> f32 output
  k_gemm_bf16<<<dim3(M / 32, P / 64), 32, 0, stream>>>(
      res_bf, W2t, b2, 1.0f, M, D, P, nullptr, out, nullptr, T, 1);
}
